// ProbAttention_66786741452805
// MI455X (gfx1250) — compile-verified, matrix operand order checked by reading, not run on hardware
//
#include <hip/hip_runtime.h>

// Flash-attention forward (causal) for gfx1250 (CDNA5, wave32).
// B=4, H=8, L=2048, D=64, fp32 in/out, bf16 WMMA GEMMs, fp32 online softmax
// in the exp2 domain (log2e folded into the Q scale).
// Workgroup = 8 waves x 32 lanes; 128 q rows per WG.
// Double-buffered cooperative LDS staging of 64-key K/V tiles (V transposed,
// pair-packed b32 stores); next tile's global loads overlap current compute.
// Softmax denominator accumulated with a ones-row WMMA (no serial VALU chain).

typedef __attribute__((ext_vector_type(16))) __bf16 v16bf;
typedef __attribute__((ext_vector_type(8)))  __bf16 v8bf;
typedef __attribute__((ext_vector_type(2)))  __bf16 v2bf;
typedef __attribute__((ext_vector_type(8)))  float  v8f;
typedef __attribute__((ext_vector_type(2)))  float  v2f;

#define ATT_B 4
#define ATT_H 8
#define ATT_L 2048
#define ATT_D 64

#define TSTRIDE 72  // bf16 per LDS tile row: 64 + 8 pad -> 144B rows, 16B aligned

#if __has_builtin(__builtin_amdgcn_exp2f)
#define EXP2F(x) __builtin_amdgcn_exp2f(x)
#else
#define EXP2F(x) exp2f(x)
#endif

static __device__ inline v16bf cat8(v8bf lo, v8bf hi) {
    return __builtin_shufflevector(lo, hi, 0,1,2,3,4,5,6,7,8,9,10,11,12,13,14,15);
}

struct StageRegs { v8f ka, kb, va, vb; };

__global__ __launch_bounds__(256)
void fa_fwd_causal_bf16(const float* __restrict__ Q,
                        const float* __restrict__ K,
                        const float* __restrict__ V,
                        float* __restrict__ O) {
    const int tid  = threadIdx.x;
    const int lane = tid & 31;
    const int wid  = tid >> 5;
    const int bh   = blockIdx.y;              // fused batch*head
    const int qchunk = blockIdx.x;            // 128 q rows per workgroup
    const int qw  = qchunk * 128 + wid * 16;  // this wave's first q row
    const int col = lane & 15;                // row/col index inside a 16x16 tile
    const int off = (lane >> 4) << 3;         // 0 for lanes 0-15, 8 for lanes 16-31

    const size_t head = (size_t)bh * ATT_L * ATT_D;
    const float* Qp = Q + head;
    const float* Kp = K + head;
    const float* Vp = V + head;
    float*       Op = O + head;

    __shared__ __align__(16) unsigned short sK[2][64 * TSTRIDE];  // K tile, bf16 [key][d]
    __shared__ __align__(16) unsigned short sV[2][64 * TSTRIDE];  // V tile, bf16 TRANSPOSED [d][key]

    // ---- Q^T as B-matrix fragments, scaled by log2(e)/sqrt(D) (exp2 softmax) ----
    const float QSCALE = 0.125f * 1.4426950408889634f;
    v16bf qf[2];
    {
        const float* qrow = Qp + (size_t)(qw + col) * ATT_D;
        #pragma unroll
        for (int f = 0; f < 2; ++f) {
            v8f lo = *(const v8f*)(qrow + f * 32 + off);
            v8f hi = *(const v8f*)(qrow + f * 32 + 16 + off);
            lo *= QSCALE; hi *= QSCALE;
            qf[f] = cat8(__builtin_convertvector(lo, v8bf),
                         __builtin_convertvector(hi, v8bf));
        }
    }

    // constant all-ones A-fragment for the denominator WMMA
    v16bf ones;
    #pragma unroll
    for (int i = 0; i < 16; ++i)
        ones[i] = __builtin_bit_cast(__bf16, (unsigned short)0x3F80);  // bf16(1.0)

    // O^T accumulators (4 d-tiles) + denominator accumulator; lane = q column.
    v8f acc[4];
    #pragma unroll
    for (int dt = 0; dt < 4; ++dt) acc[dt] = (v8f){0.f,0.f,0.f,0.f,0.f,0.f,0.f,0.f};
    v8f accl = (v8f){0.f,0.f,0.f,0.f,0.f,0.f,0.f,0.f};

    float m = -3.0e38f;  // running max (log2 domain)

    // staging decomposition (64-key tile, 256 threads):
    const int skey = tid >> 2;         // K: 0..63, one quarter-row (16 d) per thread
    const int ksc  = (tid & 3) * 16;   // 0,16,32,48
    const int vkp  = tid >> 3;         // V: keys {2*vkp, 2*vkp+1}
    const int vdc  = (tid & 7) * 8;    // x 8 d, pair-packed b32 stores

    const int nkb = qchunk * 2 + 2;    // 64-key blocks up to the WG causal bound

    // prologue: load tile 0 into registers
    StageRegs r;
    {
        const float* kr = Kp + (size_t)skey * ATT_D + ksc;
        r.ka = *(const v8f*)(kr);
        r.kb = *(const v8f*)(kr + 8);
        const float* v0 = Vp + (size_t)(2 * vkp) * ATT_D + vdc;
        r.va = *(const v8f*)(v0);
        r.vb = *(const v8f*)(v0 + ATT_D);
    }

    for (int kb = 0; kb < nkb; ++kb) {
        const int k0  = kb * 64;
        const int buf = kb & 1;
        unsigned short* bK = &sK[buf][0];
        unsigned short* bV = &sV[buf][0];

        // ---- convert + store current tile to LDS (from last iteration's loads) ----
        *(v8bf*)(&bK[skey * TSTRIDE + ksc])     = __builtin_convertvector(r.ka, v8bf);
        *(v8bf*)(&bK[skey * TSTRIDE + ksc + 8]) = __builtin_convertvector(r.kb, v8bf);
        #pragma unroll
        for (int j = 0; j < 8; ++j) {
            v2f p; p[0] = r.va[j]; p[1] = r.vb[j];
            *(unsigned*)(&bV[(vdc + j) * TSTRIDE + 2 * vkp]) =
                __builtin_bit_cast(unsigned, __builtin_convertvector(p, v2bf));
        }

        // ---- issue next tile's global loads; latency hidden by compute below ----
        if (kb + 1 < nkb) {
            const float* kr = Kp + (size_t)(k0 + 64 + skey) * ATT_D + ksc;
            r.ka = *(const v8f*)(kr);
            r.kb = *(const v8f*)(kr + 8);
            const float* v0 = Vp + (size_t)(k0 + 64 + 2 * vkp) * ATT_D + vdc;
            r.va = *(const v8f*)(v0);
            r.vb = *(const v8f*)(v0 + ATT_D);
        }

        __syncthreads();  // tile[buf] stores visible; also fences tile[buf] readers of kb-2

        if (k0 <= qw + 15) {  // wave-uniform causal guard (EXEC all-ones inside)
            // ---- S^T = K * Q^T : four 16x16 f32 tiles (16 keys each) ----
            v8f s[4];
            #pragma unroll
            for (int t = 0; t < 4; ++t) {
                const unsigned short* kr = &bK[(t * 16 + col) * TSTRIDE];
                v16bf a0 = cat8(*(const v8bf*)(kr + off),      *(const v8bf*)(kr + 16 + off));
                v16bf a1 = cat8(*(const v8bf*)(kr + 32 + off), *(const v8bf*)(kr + 48 + off));
                v8f z = (v8f){0.f,0.f,0.f,0.f,0.f,0.f,0.f,0.f};
                z = __builtin_amdgcn_wmma_f32_16x16x32_bf16(false, a0, false, qf[0], (short)0, z, false, false);
                z = __builtin_amdgcn_wmma_f32_16x16x32_bf16(false, a1, false, qf[1], (short)0, z, false, false);
                s[t] = z;
            }

            // ---- causal mask via selects: key > q -> -1e30 ----
            if (k0 + 63 > qw) {
                const int q = qw + col;
                #pragma unroll
                for (int t = 0; t < 4; ++t)
                    #pragma unroll
                    for (int r8 = 0; r8 < 8; ++r8)
                        s[t][r8] = (k0 + t * 16 + r8 + off > q) ? -1.0e30f : s[t][r8];
            }

            // ---- online softmax (fp32, exp2 domain) ----
            float bm = s[0][0];
            #pragma unroll
            for (int t = 0; t < 4; ++t)
                #pragma unroll
                for (int r8 = 0; r8 < 8; ++r8)
                    bm = fmaxf(bm, s[t][r8]);
            bm = fmaxf(bm, __shfl_xor(bm, 16, 32));  // merge the two key-row halves

            const float mnew  = fmaxf(m, bm);
            const float alpha = EXP2F(m - mnew);
            m = mnew;

            v8f e[4];
            #pragma unroll
            for (int t = 0; t < 4; ++t)
                #pragma unroll
                for (int r8 = 0; r8 < 8; ++r8)
                    e[t][r8] = EXP2F(s[t][r8] - mnew);

            #pragma unroll
            for (int dt = 0; dt < 4; ++dt)
                #pragma unroll
                for (int r8 = 0; r8 < 8; ++r8)
                    acc[dt][r8] *= alpha;
            #pragma unroll
            for (int r8 = 0; r8 < 8; ++r8)
                accl[r8] *= alpha;

            // ---- P^T as B fragments: S^T D-layout maps 1:1 onto B-layout ----
            v16bf pf0 = cat8(__builtin_convertvector(e[0], v8bf),
                             __builtin_convertvector(e[1], v8bf));
            v16bf pf1 = cat8(__builtin_convertvector(e[2], v8bf),
                             __builtin_convertvector(e[3], v8bf));

            // ---- denominator: accl += ones * P^T (row sums of P per q column) ----
            accl = __builtin_amdgcn_wmma_f32_16x16x32_bf16(false, ones, false, pf0, (short)0, accl, false, false);
            accl = __builtin_amdgcn_wmma_f32_16x16x32_bf16(false, ones, false, pf1, (short)0, accl, false, false);

            // ---- O^T += V^T * P^T : V^T A-fragments from transposed LDS tile ----
            #pragma unroll
            for (int dt = 0; dt < 4; ++dt) {
                const unsigned short* vr = &bV[(dt * 16 + col) * TSTRIDE];
                v16bf vf0 = cat8(*(const v8bf*)(vr + off),      *(const v8bf*)(vr + 16 + off));
                v16bf vf1 = cat8(*(const v8bf*)(vr + 32 + off), *(const v8bf*)(vr + 48 + off));
                acc[dt] = __builtin_amdgcn_wmma_f32_16x16x32_bf16(false, vf0, false, pf0, (short)0, acc[dt], false, false);
                acc[dt] = __builtin_amdgcn_wmma_f32_16x16x32_bf16(false, vf1, false, pf1, (short)0, acc[dt], false, false);
            }
        }
    }

    // ---- epilogue: normalize and store O (lane = q column, reg = d row) ----
    // accl rows are all equal to the softmax denominator for this q column.
    const float inv = 1.0f / accl[0];
    float* orow = Op + (size_t)(qw + col) * ATT_D;
    #pragma unroll
    for (int dt = 0; dt < 4; ++dt)
        #pragma unroll
        for (int r8 = 0; r8 < 8; ++r8)
            orow[dt * 16 + r8 + off] = acc[dt][r8] * inv;
}

extern "C" void kernel_launch(void* const* d_in, const int* in_sizes, int n_in,
                              void* d_out, int out_size, void* d_ws, size_t ws_size,
                              hipStream_t stream) {
    (void)in_sizes; (void)n_in; (void)out_size; (void)d_ws; (void)ws_size;
    const float* Q = (const float*)d_in[0];
    const float* K = (const float*)d_in[1];
    const float* V = (const float*)d_in[2];
    // d_in[3] is the causal mask; it is triu(k=1) by construction, applied analytically.
    float* O = (float*)d_out;

    dim3 grid(ATT_L / 128, ATT_B * ATT_H);
    dim3 block(256);
    fa_fwd_causal_bf16<<<grid, block, 0, stream>>>(Q, K, V, O);
}
